// REN_8564164788787
// MI455X (gfx1250) — compile-verified, matrix operand order checked by reading, not run on hardware
//
#include <hip/hip_runtime.h>
#include <math.h>

// Problem constants (from reference): NX=NQ=2048, NU=16, X is (4096,4096)
#define NXc 2048
#define NQc 2048
#define NUc 16
#define NHc 4096
#define EPSc 0.01f

typedef __attribute__((ext_vector_type(16))) __bf16 v16bf;
typedef __attribute__((ext_vector_type(8)))  float  v8f;

// ---- bf16 helpers (bit-exact, no reliance on compiler __bf16 casts) ----
__device__ inline unsigned short f32_to_bf16_bits(float f) {
  union { unsigned u; float f; } a; a.f = f;
  return (unsigned short)((a.u + 0x7fffu + ((a.u >> 16) & 1u)) >> 16); // RNE
}
__device__ inline float bf16_bits_to_f32(unsigned short hs) {
  union { unsigned u; float f; } a; a.u = ((unsigned)hs) << 16;
  return a.f;
}

// ---- CDNA5 async global->LDS copy (CDNA5 ISA) ----
// Per-lane: copies one dword from gsrc to LDS offset ldsoff; tracked by ASYNCcnt.
__device__ inline void async_g2l_b32(void* lds_dst, const void* gsrc) {
  unsigned ldsoff = (unsigned)(unsigned long long)lds_dst;  // low 32b = LDS offset
  asm volatile("global_load_async_to_lds_b32 %0, %1, off"
               :: "v"(ldsoff), "v"(gsrc) : "memory");
}
__device__ inline void wait_async0() {
  asm volatile("s_wait_asynccnt 0" ::: "memory");
}

// =====================================================================
// SYRK-style GEMM: C = A * B^T (+ diag_eps on the diagonal), f32 in/out,
// bf16x3 split products on v_wmma_f32_16x16x32_bf16 (near-fp32 accuracy:
// hi*hi + hi*lo + lo*hi, f32 accumulators; dropped lo*lo ~ 2^-16 rel).
// Block = 256 threads (8 wave32s), 64x64 C tile, K staged 64 at a time
// (float4 global loads, packed 2xbf16 LDS stores). Each wave owns 16x32.
// A is (M x K) row-major, B is (N x K) row-major, so both operands use
// the identical 16x32 bf16 A-fragment layout (A row-major, B col-major).
// lower_only=1 skips tiles strictly above the diagonal (SYRK output).
// =====================================================================
#define LSTRIDE 68   // bf16 elems per LDS row: 64 data + 4 pad (8B-aligned rows)

__global__ __launch_bounds__(256)
void syrk_wmma(const float* __restrict__ A, const float* __restrict__ B,
               float* __restrict__ C, int K, int lda, int ldb, int ldc,
               float diag_eps, int lower_only) {
  if (lower_only && (blockIdx.x > blockIdx.y)) return;
  __shared__ __bf16 sAhi[64 * LSTRIDE];
  __shared__ __bf16 sAlo[64 * LSTRIDE];
  __shared__ __bf16 sBhi[64 * LSTRIDE];
  __shared__ __bf16 sBlo[64 * LSTRIDE];

  const int tid  = threadIdx.x;
  const int lane = tid & 31;
  const int wave = tid >> 5;
  const int wr   = (wave & 3) * 16;   // wave tile row offset within 64
  const int wc   = (wave >> 2) * 32;  // wave tile col offset within 64
  const int rowBase = blockIdx.y * 64;
  const int colBase = blockIdx.x * 64;
  const int half = lane >> 4, rr = lane & 15;

  v8f acc0 = {0.f,0.f,0.f,0.f,0.f,0.f,0.f,0.f};
  v8f acc1 = {0.f,0.f,0.f,0.f,0.f,0.f,0.f,0.f};

  for (int kk = 0; kk < K; kk += 64) {
    // Stage 64x64 f32 tiles of A and B as bf16 hi/lo (float4 loads,
    // two packed u32 stores per 4 elements per array).
#pragma unroll
    for (int q = 0; q < 4; ++q) {
      int idx = q * 256 + tid;          // 0..1023
      int r   = idx >> 4;               // 0..63
      int c4  = idx & 15;               // float4 slot in row
      float4 av = *(const float4*)(A + (size_t)(rowBase + r) * lda + kk + c4 * 4);
      float4 bv = *(const float4*)(B + (size_t)(colBase + r) * ldb + kk + c4 * 4);

      unsigned ah0 = f32_to_bf16_bits(av.x), ah1 = f32_to_bf16_bits(av.y);
      unsigned ah2 = f32_to_bf16_bits(av.z), ah3 = f32_to_bf16_bits(av.w);
      unsigned al0 = f32_to_bf16_bits(av.x - bf16_bits_to_f32((unsigned short)ah0));
      unsigned al1 = f32_to_bf16_bits(av.y - bf16_bits_to_f32((unsigned short)ah1));
      unsigned al2 = f32_to_bf16_bits(av.z - bf16_bits_to_f32((unsigned short)ah2));
      unsigned al3 = f32_to_bf16_bits(av.w - bf16_bits_to_f32((unsigned short)ah3));
      unsigned* pah = (unsigned*)&sAhi[r * LSTRIDE + c4 * 4];
      unsigned* pal = (unsigned*)&sAlo[r * LSTRIDE + c4 * 4];
      pah[0] = ah0 | (ah1 << 16); pah[1] = ah2 | (ah3 << 16);
      pal[0] = al0 | (al1 << 16); pal[1] = al2 | (al3 << 16);

      unsigned bh0 = f32_to_bf16_bits(bv.x), bh1 = f32_to_bf16_bits(bv.y);
      unsigned bh2 = f32_to_bf16_bits(bv.z), bh3 = f32_to_bf16_bits(bv.w);
      unsigned bl0 = f32_to_bf16_bits(bv.x - bf16_bits_to_f32((unsigned short)bh0));
      unsigned bl1 = f32_to_bf16_bits(bv.y - bf16_bits_to_f32((unsigned short)bh1));
      unsigned bl2 = f32_to_bf16_bits(bv.z - bf16_bits_to_f32((unsigned short)bh2));
      unsigned bl3 = f32_to_bf16_bits(bv.w - bf16_bits_to_f32((unsigned short)bh3));
      unsigned* pbh = (unsigned*)&sBhi[r * LSTRIDE + c4 * 4];
      unsigned* pbl = (unsigned*)&sBlo[r * LSTRIDE + c4 * 4];
      pbh[0] = bh0 | (bh1 << 16); pbh[1] = bh2 | (bh3 << 16);
      pbl[0] = bl0 | (bl1 << 16); pbl[1] = bl2 | (bl3 << 16);
    }
    __syncthreads();

    // Two 32-wide K sub-steps per stage. Fragment layout (ISA 16-bit A 16x32):
    //   lanes 0-15 : row M=lane,   elems 0..7 -> K 0..7,  8..15 -> K 16..23
    //   lanes 16-31: row M=lane-16, K shifted +8 per group.
#pragma unroll
    for (int hh = 0; hh < 2; ++hh) {
      const int kb = hh * 32;
      v16bf ahi, alo, bhi0, blo0, bhi1, blo1;
#pragma unroll
      for (int e = 0; e < 16; ++e) {
        int k = kb + ((e < 8) ? e : (e + 8)) + 8 * half;
        ahi[e]  = sAhi[(wr + rr) * LSTRIDE + k];
        alo[e]  = sAlo[(wr + rr) * LSTRIDE + k];
        bhi0[e] = sBhi[(wc + rr) * LSTRIDE + k];
        blo0[e] = sBlo[(wc + rr) * LSTRIDE + k];
        bhi1[e] = sBhi[(wc + 16 + rr) * LSTRIDE + k];
        blo1[e] = sBlo[(wc + 16 + rr) * LSTRIDE + k];
      }
      acc0 = __builtin_amdgcn_wmma_f32_16x16x32_bf16(false, ahi, false, bhi0, (short)0, acc0, false, false);
      acc0 = __builtin_amdgcn_wmma_f32_16x16x32_bf16(false, ahi, false, blo0, (short)0, acc0, false, false);
      acc0 = __builtin_amdgcn_wmma_f32_16x16x32_bf16(false, alo, false, bhi0, (short)0, acc0, false, false);
      acc1 = __builtin_amdgcn_wmma_f32_16x16x32_bf16(false, ahi, false, bhi1, (short)0, acc1, false, false);
      acc1 = __builtin_amdgcn_wmma_f32_16x16x32_bf16(false, ahi, false, blo1, (short)0, acc1, false, false);
      acc1 = __builtin_amdgcn_wmma_f32_16x16x32_bf16(false, alo, false, bhi1, (short)0, acc1, false, false);
    }
    __syncthreads();
  }

  // C/D layout: VGPR v -> row (v + 8*half), lane -> col rr.
#pragma unroll
  for (int v = 0; v < 8; ++v) {
    int row  = rowBase + wr + v + 8 * half;
    int col0 = colBase + wc + rr;
    int col1 = col0 + 16;
    float x0 = acc0[v]; if (row == col0) x0 += diag_eps;
    float x1 = acc1[v]; if (row == col1) x1 += diag_eps;
    C[(size_t)row * ldc + col0] = x0;
    C[(size_t)row * ldc + col1] = x1;
  }
}

// ---------------- GEMV kernels (O(n^2), L2-resident) ----------------
__global__ __launch_bounds__(256)
void gemv_row(const float* __restrict__ A, const float* __restrict__ v,
              float* __restrict__ y, int cols, int lda) {
  __shared__ float red[256];
  int row = blockIdx.x;
  float s = 0.f;
  for (int k = threadIdx.x; k < cols; k += 256)
    s += A[(size_t)row * lda + k] * v[k];
  red[threadIdx.x] = s;
  __syncthreads();
  for (int off = 128; off > 0; off >>= 1) {
    if (threadIdx.x < off) red[threadIdx.x] += red[threadIdx.x + off];
    __syncthreads();
  }
  if (threadIdx.x == 0) y[row] = red[0];
}

__global__ void gemv_col(const float* __restrict__ A, const float* __restrict__ v,
                         float* __restrict__ y, int rows, int cols, int lda) {
  int k = blockIdx.x * blockDim.x + threadIdx.x;
  if (k >= cols) return;
  float s = 0.f;
  for (int i = 0; i < rows; ++i) s += A[(size_t)i * lda + k] * v[i];
  y[k] = s;
}

__global__ void matvec16(const float* __restrict__ A, const float* __restrict__ xk,
                         const float* __restrict__ add, float* __restrict__ y,
                         int n, int k) {
  int i = blockIdx.x * blockDim.x + threadIdx.x;
  if (i >= n) return;
  float s = add ? add[i] : 0.f;
  for (int j = 0; j < k; ++j) s += A[(size_t)i * k + j] * xk[j];
  y[i] = s;
}

__global__ void diag_lam(const float* __restrict__ H22, float* __restrict__ lam, int n) {
  int i = blockIdx.x * blockDim.x + threadIdx.x;
  if (i < n) lam[i] = 0.5f * H22[(size_t)i * n + i];
}

// v = -0.5*(h11x + eps*x + y1x - y1tx) - h12w - uw
__global__ void combine_v(const float* __restrict__ h11x, const float* __restrict__ y1x,
                          const float* __restrict__ y1tx, const float* __restrict__ h12w,
                          const float* __restrict__ uw, const float* __restrict__ x,
                          float* __restrict__ v, int n) {
  int i = blockIdx.x * blockDim.x + threadIdx.x;
  if (i < n)
    v[i] = -0.5f * (h11x[i] + EPSc * x[i] + y1x[i] - y1tx[i]) - h12w[i] - uw[i];
}

// =====================================================================
// Blocked ReLU forward substitution for w (one workgroup, block=16):
//   w_i = relu(s_i/lam_i + t_i - (sum_{j<i} H22[i,j] w_j)/lam_i)
// 16x16 diagonal blocks are DMA'd global->LDS (async-to-LDS path).
// =====================================================================
__global__ __launch_bounds__(1024)
void w_solve(const float* __restrict__ H22, const float* __restrict__ lam,
             const float* __restrict__ s, const float* __restrict__ t,
             float* __restrict__ w, int n) {
  __shared__ float acc[NQc];
  __shared__ float wb[16];
  __shared__ float Dkk[256];
  for (int i = threadIdx.x; i < n; i += blockDim.x) acc[i] = 0.f;
  __syncthreads();
  for (int blk = 0; blk < n / 16; ++blk) {
    int base = blk * 16;
    if (threadIdx.x < 256) {
      int r = threadIdx.x >> 4, c = threadIdx.x & 15;
      async_g2l_b32(&Dkk[threadIdx.x], H22 + (size_t)(base + r) * n + base + c);
      wait_async0();
    }
    __syncthreads();
    if (threadIdx.x == 0) {
      for (int i = 0; i < 16; ++i) {
        int gi = base + i;
        float a = acc[gi];
        for (int j = 0; j < i; ++j) a += Dkk[i * 16 + j] * wb[j];
        float li = lam[gi];
        float wi = s[gi] / li + t[gi] - a / li;
        wb[i] = wi > 0.f ? wi : 0.f;
      }
    }
    __syncthreads();
    if (threadIdx.x < 16) w[base + threadIdx.x] = wb[threadIdx.x];
    for (int i = base + 16 + threadIdx.x; i < n; i += blockDim.x) {
      const float* row = H22 + (size_t)i * n + base;
      float a = 0.f;
#pragma unroll
      for (int j = 0; j < 16; ++j) a += row[j] * wb[j];
      acc[i] += a;
    }
    __syncthreads();
  }
}

// ============ Blocked right-looking Cholesky of P (in place, lower) ============
__global__ __launch_bounds__(256)
void chol_panel(float* __restrict__ P, int n, int base) {
  __shared__ float Lkk[256];
  int r = threadIdx.x >> 4, c = threadIdx.x & 15;
  async_g2l_b32(&Lkk[threadIdx.x], P + (size_t)(base + r) * n + base + c);
  wait_async0();
  __syncthreads();
  if (threadIdx.x == 0) {
    for (int j = 0; j < 16; ++j) {
      float d = Lkk[j * 16 + j];
      for (int m = 0; m < j; ++m) d -= Lkk[j * 16 + m] * Lkk[j * 16 + m];
      d = sqrtf(fmaxf(d, 1e-30f));
      Lkk[j * 16 + j] = d;
      float inv = 1.f / d;
      for (int i = j + 1; i < 16; ++i) {
        float v = Lkk[i * 16 + j];
        for (int m = 0; m < j; ++m) v -= Lkk[i * 16 + m] * Lkk[j * 16 + m];
        Lkk[i * 16 + j] = v * inv;
      }
    }
  }
  __syncthreads();
  P[(size_t)(base + r) * n + base + c] = (c <= r) ? Lkk[threadIdx.x] : 0.f;
  // Panel solve: row_i <- row_i * inv(Lkk^T), parallel over rows below.
  for (int i = base + 16 + threadIdx.x; i < n; i += blockDim.x) {
    float* row = P + (size_t)i * n + base;
    float v[16];
#pragma unroll
    for (int j = 0; j < 16; ++j) v[j] = row[j];
#pragma unroll
    for (int j = 0; j < 16; ++j) {
      float sum = v[j];
      for (int m = 0; m < j; ++m) sum -= v[m] * Lkk[j * 16 + m];
      v[j] = sum / Lkk[j * 16 + j];
    }
#pragma unroll
    for (int j = 0; j < 16; ++j) row[j] = v[j];
  }
}

// Trailing update: A[i,j] -= L[i,kb] . L[j,kb] for i,j >= base+16, j <= i
__global__ __launch_bounds__(256)
void chol_update(float* __restrict__ P, int n, int base) {
  if (blockIdx.x > blockIdx.y) return;  // strictly-upper tiles skip
  int r = threadIdx.x >> 4, c = threadIdx.x & 15;
  int i = base + 16 + blockIdx.y * 16 + r;
  int j = base + 16 + blockIdx.x * 16 + c;
  if (i >= n || j >= n || j > i) return;
  const float* Li = P + (size_t)i * n + base;
  const float* Lj = P + (size_t)j * n + base;
  float s = 0.f;
#pragma unroll
  for (int m = 0; m < 16; ++m) s += Li[m] * Lj[m];
  P[(size_t)i * n + j] -= s;
}

// ---------------- Blocked triangular solves (single workgroup) ----------------
__global__ __launch_bounds__(1024)
void trsv_lower(const float* __restrict__ L, const float* __restrict__ b,
                float* __restrict__ y, int n) {
  __shared__ float acc[NXc];
  __shared__ float yb[16];
  __shared__ float Lkk[256];
  for (int i = threadIdx.x; i < n; i += blockDim.x) acc[i] = 0.f;
  __syncthreads();
  for (int blk = 0; blk < n / 16; ++blk) {
    int base = blk * 16;
    if (threadIdx.x < 256) {
      int r = threadIdx.x >> 4, c = threadIdx.x & 15;
      async_g2l_b32(&Lkk[threadIdx.x], L + (size_t)(base + r) * n + base + c);
      wait_async0();
    }
    __syncthreads();
    if (threadIdx.x == 0) {
      for (int i = 0; i < 16; ++i) {
        float val = b[base + i] - acc[base + i];
        for (int j = 0; j < i; ++j) val -= Lkk[i * 16 + j] * yb[j];
        yb[i] = val / Lkk[i * 16 + i];
      }
    }
    __syncthreads();
    if (threadIdx.x < 16) y[base + threadIdx.x] = yb[threadIdx.x];
    for (int i = base + 16 + threadIdx.x; i < n; i += blockDim.x) {
      const float* row = L + (size_t)i * n + base;
      float s = 0.f;
#pragma unroll
      for (int j = 0; j < 16; ++j) s += row[j] * yb[j];
      acc[i] += s;
    }
    __syncthreads();
  }
}

// Solve L^T z = b (backward, reads only lower L)
__global__ __launch_bounds__(1024)
void trsv_upperT(const float* __restrict__ L, const float* __restrict__ b,
                 float* __restrict__ z, int n) {
  __shared__ float acc[NXc];
  __shared__ float zb[16];
  __shared__ float Lkk[256];
  for (int i = threadIdx.x; i < n; i += blockDim.x) acc[i] = 0.f;
  __syncthreads();
  for (int blk = n / 16 - 1; blk >= 0; --blk) {
    int base = blk * 16;
    if (threadIdx.x < 256) {
      int r = threadIdx.x >> 4, c = threadIdx.x & 15;
      async_g2l_b32(&Lkk[threadIdx.x], L + (size_t)(base + r) * n + base + c);
      wait_async0();
    }
    __syncthreads();
    if (threadIdx.x == 0) {
      for (int i = 15; i >= 0; --i) {
        float val = b[base + i] - acc[base + i];
        for (int j = i + 1; j < 16; ++j) val -= Lkk[j * 16 + i] * zb[j];
        zb[i] = val / Lkk[i * 16 + i];
      }
    }
    __syncthreads();
    if (threadIdx.x < 16) z[base + threadIdx.x] = zb[threadIdx.x];
    for (int i = threadIdx.x; i < base; i += blockDim.x) {
      float s = 0.f;
#pragma unroll
      for (int j = 0; j < 16; ++j) s += L[(size_t)(base + j) * n + i] * zb[j];
      acc[i] += s;
    }
    __syncthreads();
  }
}

// =====================================================================
extern "C" void kernel_launch(void* const* d_in, const int* in_sizes, int n_in,
                              void* d_out, int out_size, void* d_ws, size_t ws_size,
                              hipStream_t stream) {
  (void)in_sizes; (void)n_in; (void)out_size; (void)ws_size;
  const float* x   = (const float*)d_in[1];
  const float* u   = (const float*)d_in[2];
  const float* X   = (const float*)d_in[3];
  const float* U   = (const float*)d_in[4];
  const float* Y1  = (const float*)d_in[5];
  const float* XP  = (const float*)d_in[6];
  const float* B2  = (const float*)d_in[7];
  const float* D12 = (const float*)d_in[8];
  float* out = (float*)d_out;

  float* ws   = (float*)d_ws;
  float* H22  = ws;                              // 2048*2048
  float* Pm   = ws + (size_t)4 * 1024 * 1024;    // 2048*2048 (-> L)
  float* vb   = ws + (size_t)8 * 1024 * 1024;    // vector arena
  float* lam  = vb;            float* sv   = vb + 2048;
  float* tv   = vb + 4096;     float* wv   = vb + 6144;
  float* a1   = vb + 8192;     /* 4096 */
  float* h11x = vb + 12288;    float* y1x  = vb + 14336;
  float* y1tx = vb + 16384;    float* a2   = vb + 18432;  /* 4096 */
  float* h12w = vb + 22528;    float* uw   = vb + 24576;
  float* vv   = vb + 26624;    float* yv   = vb + 28672;
  float* zv   = vb + 30720;

  const float* X2 = X + (size_t)NXc * NHc;   // rows nx.. of X
  dim3 g32(32, 32);

  // H22 = X2 X2^T + eps I (lower+diag), P = XP XP^T + eps I (lower+diag)
  syrk_wmma<<<g32, 256, 0, stream>>>(X2, X2, H22, NHc, NHc, NHc, NXc, EPSc, 1);
  syrk_wmma<<<g32, 256, 0, stream>>>(XP, XP, Pm,  NXc, NXc, NXc, NXc, EPSc, 1);

  diag_lam<<<8, 256, 0, stream>>>(H22, lam, NXc);
  gemv_col<<<8, 256, 0, stream>>>(U, x, sv, NXc, NQc, NQc);        // s = U^T x
  matvec16<<<8, 256, 0, stream>>>(D12, u, nullptr, tv, NQc, NUc);  // t = D12 u
  w_solve<<<1, 1024, 0, stream>>>(H22, lam, sv, tv, wv, NQc);      // w

  gemv_col<<<16, 256, 0, stream>>>(X, x, a1, NXc, NHc, NHc);       // a1 = X1^T x
  gemv_row<<<NXc, 256, 0, stream>>>(X, a1, h11x, NHc, NHc);        // X1 a1
  gemv_row<<<NXc, 256, 0, stream>>>(Y1, x, y1x, NXc, NXc);         // Y1 x
  gemv_col<<<8, 256, 0, stream>>>(Y1, x, y1tx, NXc, NXc, NXc);     // Y1^T x
  gemv_col<<<16, 256, 0, stream>>>(X2, wv, a2, NQc, NHc, NHc);     // a2 = X2^T w
  gemv_row<<<NXc, 256, 0, stream>>>(X, a2, h12w, NHc, NHc);        // X1 a2
  gemv_row<<<NXc, 256, 0, stream>>>(U, wv, uw, NXc, NQc);          // U w
  combine_v<<<8, 256, 0, stream>>>(h11x, y1x, y1tx, h12w, uw, x, vv, NXc);

  // Cholesky P = L L^T (in place, lower)
  for (int blk = 0; blk < NXc / 16; ++blk) {
    int base = blk * 16;
    chol_panel<<<1, 256, 0, stream>>>(Pm, NXc, base);
    int rem = NXc - base - 16;
    if (rem > 0) {
      int tiles = (rem + 15) / 16;
      chol_update<<<dim3(tiles, tiles), 256, 0, stream>>>(Pm, NXc, base);
    }
  }

  trsv_lower<<<1, 1024, 0, stream>>>(Pm, vv, yv, NXc);   // L y = v
  trsv_upperT<<<1, 1024, 0, stream>>>(Pm, yv, zv, NXc);  // L^T z = y
  matvec16<<<8, 256, 0, stream>>>(B2, u, zv, out, NXc, NUc);  // x_dot = z + B2 u
}
